// MHA_70514773066282
// MI455X (gfx1250) — matrix operand order repaired and checked by reading, not compiled, not run on hardware
//
#include <hip/hip_runtime.h>
#include <hip/hip_bf16.h>

typedef __attribute__((ext_vector_type(16))) _Float16 v16h;
typedef __attribute__((ext_vector_type(8)))  float    v8f;

#define DEVI __device__ __forceinline__

// ---------------- problem constants ----------------
static constexpr int N_EMBD = 576, N_HEAD = 9, N_KV = 3, HEAD_DIM = 64;
static constexpr int DR = 8, GQA = 3, DQ = HEAD_DIM - DR;          // 56
static constexpr int Bb = 2, Ss = 2064, Mrows = Bb * Ss;           // 4128
static constexpr int NPROJ = N_HEAD*DQ + N_KV*DQ + N_KV*HEAD_DIM + N_HEAD*DR + N_KV*DR; // 960
static constexpr int OFF_Q = 0, OFF_K = 504, OFF_V = 672, OFF_QR = 864, OFF_KR = 936;
static constexpr float ROPE_THETA = 100000.0f, CLIPV = 10000.0f;
static constexpr int GROUPQ = 64, RESID = 32;
static constexpr int KQ_LEN = (Ss / RESID) * RESID;                // 2048
static constexpr int VQ_LEN = Ss - RESID;                          // 2032

// ---------------- workspace layout (bytes) ----------------
static constexpr size_t O_XH   = 0;                                  // 4128*576 f16
static constexpr size_t O_WCAT = O_XH   + (size_t)Mrows*N_EMBD*2;    // 960*576 f16
static constexpr size_t O_OWH  = O_WCAT + (size_t)NPROJ*N_EMBD*2;    // 576*576 f16
static constexpr size_t O_PROJ = O_OWH  + (size_t)N_EMBD*N_EMBD*2;   // 4128*960 f32
static constexpr size_t O_QB   = O_PROJ + (size_t)Mrows*NPROJ*4;     // B*H*S*64 f16
static constexpr size_t O_KB   = O_QB   + (size_t)Bb*N_HEAD*Ss*64*2; // B*KV*S*64 f16
static constexpr size_t O_VT   = O_KB   + (size_t)Bb*N_KV*Ss*64*2;   // B*KV*64*S f16 (transposed)
static constexpr size_t O_CTX  = O_VT   + (size_t)Bb*N_KV*Ss*64*2;   // B*S*576 f16

// ---------------- WMMA fragment loaders (CDNA5 wave32 layouts) ----------------
// 16-bit A-matrix 16x32: lanes 0-15 -> M=lane, K pairs {0..7,16..23};
//                        lanes 16-31 -> M=lane-16, K pairs {8..15,24..31}
DEVI v16h load_frag_a(const _Float16* p, int ld) {
  int lane = threadIdx.x & 31, m = lane & 15, hi = lane >> 4;
  union { v16h v; unsigned u[8]; } f;
#pragma unroll
  for (int r = 0; r < 8; ++r) {
    int k = 2*r + (r >= 4 ? 8 : 0) + 8*hi;
    f.u[r] = *(const unsigned*)(p + m*ld + k);
  }
  return f.v;
}
// 16-bit B-matrix 32x16 from [n][k] storage: lanes 0-15 -> N=lane, K {0..7,16..23};
//                                            lanes 16-31 -> N=lane-16, K {8..15,24..31}
DEVI v16h load_frag_b(const _Float16* p, int ld) {
  int lane = threadIdx.x & 31, n = lane & 15, hi = lane >> 4;
  union { v16h v; unsigned u[8]; } f;
#pragma unroll
  for (int r = 0; r < 8; ++r) {
    int k = 2*r + (r >= 4 ? 8 : 0) + 8*hi;
    f.u[r] = *(const unsigned*)(p + n*ld + k);
  }
  return f.v;
}

// ---------------- kernel 1: pack inputs to f16 ----------------
__global__ void pack_f16(const float* __restrict__ x,  const float* __restrict__ qw,
                         const float* __restrict__ kw, const float* __restrict__ vw,
                         const float* __restrict__ qrw,const float* __restrict__ krw,
                         const float* __restrict__ ow,
                         _Float16* __restrict__ xh, _Float16* __restrict__ wcat,
                         _Float16* __restrict__ owh) {
  const int TX = Mrows*N_EMBD, TW = NPROJ*N_EMBD, TO = N_EMBD*N_EMBD;
  for (int i = blockIdx.x*blockDim.x + threadIdx.x; i < TX+TW+TO;
       i += gridDim.x*blockDim.x) {
    if (i < TX) { xh[i] = (_Float16)x[i]; }
    else if (i < TX + TW) {
      int j = i - TX, n = j / N_EMBD, c = j % N_EMBD;
      const float* src; int r;
      if      (n < OFF_K)  { src = qw;  r = n; }
      else if (n < OFF_V)  { src = kw;  r = n - OFF_K; }
      else if (n < OFF_QR) { src = vw;  r = n - OFF_V; }
      else if (n < OFF_KR) { src = qrw; r = n - OFF_QR; }
      else                 { src = krw; r = n - OFF_KR; }
      wcat[j] = (_Float16)src[(size_t)r*N_EMBD + c];
    } else {
      int j = i - TX - TW;
      owh[j] = (_Float16)ow[j];
    }
  }
}

// ---------------- kernel 2: generic WMMA GEMM  C(MxN,f32) = A(MxK,f16) * B(NxK,f16)^T
__global__ __launch_bounds__(128) void gemm_wmma_f16(
    const _Float16* __restrict__ A, const _Float16* __restrict__ Bw,
    float* __restrict__ C, int M, int N, int K) {
  __shared__ _Float16 sA[64*32];
  __shared__ _Float16 sB[64*32];
  const int tid = threadIdx.x, wave = tid >> 5, lane = tid & 31;
  const int ln = lane & 15, hi = lane >> 4;
  const int mBase = blockIdx.y * 64, nBase = blockIdx.x * 64;
  const int row = tid >> 1, seg = tid & 1;

  const v8f vzero = {};
  v8f acc[4] = {vzero, vzero, vzero, vzero};

  for (int kc = 0; kc < K; kc += 32) {
    __syncthreads();
    {
      uint4 a0 = make_uint4(0,0,0,0), a1 = make_uint4(0,0,0,0);
      int gr = mBase + row;
      if (gr < M) {
        const uint4* src = (const uint4*)(A + (size_t)gr*K + kc + seg*16);
        a0 = src[0]; a1 = src[1];
        __builtin_prefetch((const void*)(A + (size_t)gr*K + kc + 64), 0, 1);
      }
      uint4* dst = (uint4*)(sA + row*32 + seg*16);
      dst[0] = a0; dst[1] = a1;
    }
    {
      const uint4* src = (const uint4*)(Bw + (size_t)(nBase + row)*K + kc + seg*16);
      uint4* dst = (uint4*)(sB + row*32 + seg*16);
      dst[0] = src[0]; dst[1] = src[1];
    }
    __syncthreads();

    v16h aF = load_frag_a(sA + wave*16*32, 32);
#pragma unroll
    for (int t = 0; t < 4; ++t) {
      v16h bF = load_frag_b(sB + t*16*32, 32);
      acc[t] = __builtin_amdgcn_wmma_f32_16x16x32_f16(false, aF, false, bF,
                                                      (short)0, acc[t], false, false);
    }
  }
#pragma unroll
  for (int t = 0; t < 4; ++t)
#pragma unroll
    for (int r = 0; r < 8; ++r) {
      int gm = mBase + wave*16 + r + 8*hi;
      if (gm < M) C[(size_t)gm*N + nBase + t*16 + ln] = acc[t][r];
    }
}

// ---------------- RoPE helper (matches reference idx selection) ----------------
DEVI float rope_t(int j4) {
  float e = (j4 == 0) ? 0.f : (j4 == 1) ? (2.f/64.f) : (j4 == 2) ? (32.f/64.f) : (34.f/64.f);
  return powf(ROPE_THETA, -e);
}
DEVI float k_raw(const float* pr, int kv, int d, int s) {
  if (d < DQ) return pr[OFF_K + kv*DQ + d];
  int j = d - DQ;
  float xj = pr[OFF_KR + kv*DR + j];
  float pv = (j < 4) ? -pr[OFF_KR + kv*DR + j + 4] : pr[OFF_KR + kv*DR + j - 4];
  float a = (float)s * rope_t(j & 3);
  return xj*cosf(a) + pv*sinf(a);
}

// ---------------- kernel 3: assemble Q (nope||rope) ----------------
__global__ void finish_q(const float* __restrict__ proj, _Float16* __restrict__ Q) {
  const int total = Bb*N_HEAD*Ss*HEAD_DIM;
  for (int i = blockIdx.x*blockDim.x + threadIdx.x; i < total;
       i += gridDim.x*blockDim.x) {
    int d = i & 63, s = (i >> 6) % Ss, h = (i/(64*Ss)) % N_HEAD, b = i/(64*Ss*N_HEAD);
    const float* pr = proj + (size_t)(b*Ss + s)*NPROJ;
    float val;
    if (d < DQ) val = pr[OFF_Q + h*DQ + d];
    else {
      int j = d - DQ;
      float xj = pr[OFF_QR + h*DR + j];
      float pv = (j < 4) ? -pr[OFF_QR + h*DR + j + 4] : pr[OFF_QR + h*DR + j - 4];
      float a = (float)s * rope_t(j & 3);
      val = xj*cosf(a) + pv*sinf(a);
    }
    Q[i] = (_Float16)val;
  }
}

// ---------------- kernel 4: assemble + 2-bit fake-quant K ----------------
__global__ __launch_bounds__(64) void finish_k(const float* __restrict__ proj,
                                               _Float16* __restrict__ Kb) {
  const int g = blockIdx.x, kv = blockIdx.y, b = blockIdx.z, d = threadIdx.x;
  const int s0 = g * GROUPQ;
  if (g < KQ_LEN / GROUPQ) {                       // quantized groups of 64 positions
    float mn = 1e30f, mx = -1e30f;
    for (int p = 0; p < GROUPQ; ++p) {
      int s = s0 + p;
      float v = k_raw(proj + (size_t)(b*Ss + s)*NPROJ, kv, d, s);
      v = fminf(fmaxf(v, -CLIPV), CLIPV);
      mn = fminf(mn, v); mx = fmaxf(mx, v);
    }
    float sc = (mx - mn) * (1.f/3.f);
    if (sc == 0.f) sc = 1.f;
    for (int p = 0; p < GROUPQ; ++p) {
      int s = s0 + p;
      float v = k_raw(proj + (size_t)(b*Ss + s)*NPROJ, kv, d, s);
      v = fminf(fmaxf(v, -CLIPV), CLIPV);
      float deq = rintf((v - mn)/sc)*sc + mn;
      Kb[((size_t)(b*N_KV + kv)*Ss + s)*HEAD_DIM + d] = (_Float16)deq;
    }
  } else {                                         // residual tail, unquantized
    for (int s = KQ_LEN; s < Ss; ++s) {
      float v = k_raw(proj + (size_t)(b*Ss + s)*NPROJ, kv, d, s);
      Kb[((size_t)(b*N_KV + kv)*Ss + s)*HEAD_DIM + d] = (_Float16)v;
    }
  }
}

// ---------------- kernel 5: 2-bit fake-quant V, store transposed [d][s] ----------------
__global__ void finish_v(const float* __restrict__ proj, _Float16* __restrict__ VT) {
  const int totalRows = Bb*N_KV*Ss;
  for (int rr = blockIdx.x*blockDim.x + threadIdx.x; rr < totalRows;
       rr += gridDim.x*blockDim.x) {
    int s = rr % Ss, kv = (rr/Ss) % N_KV, b = rr/(Ss*N_KV);
    const float* pr = proj + (size_t)(b*Ss + s)*NPROJ + OFF_V + kv*HEAD_DIM;
    size_t base = (size_t)(b*N_KV + kv)*HEAD_DIM;
    if (s < VQ_LEN) {
      float mn = 1e30f, mx = -1e30f;
      for (int d = 0; d < HEAD_DIM; ++d) {
        float v = fminf(fmaxf(pr[d], -CLIPV), CLIPV);
        mn = fminf(mn, v); mx = fmaxf(mx, v);
      }
      float sc = (mx - mn) * (1.f/3.f);
      if (sc == 0.f) sc = 1.f;
      for (int d = 0; d < HEAD_DIM; ++d) {
        float v = fminf(fmaxf(pr[d], -CLIPV), CLIPV);
        float deq = rintf((v - mn)/sc)*sc + mn;
        VT[(base + d)*Ss + s] = (_Float16)deq;
      }
    } else {
      for (int d = 0; d < HEAD_DIM; ++d) VT[(base + d)*Ss + s] = (_Float16)pr[d];
    }
  }
}

// ---------------- kernel 6: flash attention with WMMA ----------------
__global__ __launch_bounds__(128) void attn_wmma(
    const _Float16* __restrict__ Q, const _Float16* __restrict__ Kb,
    const _Float16* __restrict__ VT, _Float16* __restrict__ ctx) {
  __shared__ _Float16 sQ[64*64];
  __shared__ _Float16 sK[32*64];
  __shared__ _Float16 sVT[64*32];
  __shared__ _Float16 sP[4*16*32];

  const int tid = threadIdx.x, wave = tid >> 5, lane = tid & 31;
  const int ln = lane & 15, hi = lane >> 4;
  const int qtile = blockIdx.x, bh = blockIdx.y;
  const int b = bh / N_HEAD, h = bh % N_HEAD, kv = h / GQA;
  const int qb = qtile * 64;
  const size_t qBase = (size_t)bh * Ss;
  const size_t kBase = (size_t)(b*N_KV + kv) * Ss;
  const size_t vBase = (size_t)(b*N_KV + kv) * HEAD_DIM;

  { // load 64x64 Q tile
    int row = tid >> 1, seg = tid & 1;
    uint4 z = make_uint4(0,0,0,0), v0 = z, v1 = z, v2 = z, v3 = z;
    if (qb + row < Ss) {
      const uint4* src = (const uint4*)(Q + (qBase + qb + row)*64 + seg*32);
      v0 = src[0]; v1 = src[1]; v2 = src[2]; v3 = src[3];
    }
    uint4* dst = (uint4*)(sQ + row*64 + seg*32);
    dst[0] = v0; dst[1] = v1; dst[2] = v2; dst[3] = v3;
  }
  __syncthreads();
  const v16h aQ0 = load_frag_a(sQ + wave*16*64,      64);
  const v16h aQ1 = load_frag_a(sQ + wave*16*64 + 32, 64);

  const v8f vzero = {};
  v8f acc[4] = {vzero, vzero, vzero, vzero};
  float m_st[8], l_st[8];
#pragma unroll
  for (int r = 0; r < 8; ++r) { m_st[r] = -1e30f; l_st[r] = 0.f; }

  const float scl = 0.125f;                     // 1/sqrt(64)
  int kbEnd = qb + 64; if (kbEnd > Ss) kbEnd = Ss;

  for (int kb = 0; kb < kbEnd; kb += 32) {
    __syncthreads();
    { // K block: 32 keys x 64 dims
      int row = tid >> 2, seg = tid & 3;
      uint4 z = make_uint4(0,0,0,0), v0 = z, v1 = z;
      if (kb + row < Ss) {
        const uint4* src = (const uint4*)(Kb + (kBase + kb + row)*64 + seg*16);
        v0 = src[0]; v1 = src[1];
      }
      uint4* dst = (uint4*)(sK + row*64 + seg*16);
      dst[0] = v0; dst[1] = v1;
    }
    { // V^T block: 64 dims x 32 keys
      int row = tid >> 1, seg = tid & 1;
      const _Float16* src = VT + (vBase + row)*Ss + kb + seg*16;
      if (kb + 32 <= Ss) {
        uint4* dst = (uint4*)(sVT + row*32 + seg*16);
        dst[0] = ((const uint4*)src)[0];
        dst[1] = ((const uint4*)src)[1];
      } else {
        for (int i = 0; i < 16; ++i) {
          int key = kb + seg*16 + i;
          sVT[row*32 + seg*16 + i] = (key < Ss) ? src[i] : (_Float16)0.f;
        }
      }
    }
    __syncthreads();

    // scores: 16 queries x 32 keys (two 16x16 tiles)
    v8f c0 = vzero, c1 = vzero;
    {
      v16h b00 = load_frag_b(sK,            64);
      v16h b01 = load_frag_b(sK + 32,       64);
      c0 = __builtin_amdgcn_wmma_f32_16x16x32_f16(false, aQ0, false, b00, (short)0, c0, false, false);
      c0 = __builtin_amdgcn_wmma_f32_16x16x32_f16(false, aQ1, false, b01, (short)0, c0, false, false);
      v16h b10 = load_frag_b(sK + 16*64,      64);
      v16h b11 = load_frag_b(sK + 16*64 + 32, 64);
      c1 = __builtin_amdgcn_wmma_f32_16x16x32_f16(false, aQ0, false, b10, (short)0, c1, false, false);
      c1 = __builtin_amdgcn_wmma_f32_16x16x32_f16(false, aQ1, false, b11, (short)0, c1, false, false);
    }

    // online softmax per row (row = r + 8*hi of C-fragment layout)
#pragma unroll
    for (int r = 0; r < 8; ++r) {
      int qrow = qb + wave*16 + r + 8*hi;
      int col0 = kb + ln, col1 = kb + 16 + ln;
      float s0v = c0[r]*scl, s1v = c1[r]*scl;
      if (col0 > qrow || col0 >= Ss || qrow >= Ss) s0v = -1e30f;
      if (col1 > qrow || col1 >= Ss || qrow >= Ss) s1v = -1e30f;
      float rm = fmaxf(s0v, s1v);
      rm = fmaxf(rm, __shfl_xor(rm, 1, 32));
      rm = fmaxf(rm, __shfl_xor(rm, 2, 32));
      rm = fmaxf(rm, __shfl_xor(rm, 4, 32));
      rm = fmaxf(rm, __shfl_xor(rm, 8, 32));
      float m_new = fmaxf(m_st[r], rm);
      float alpha = __expf(m_st[r] - m_new);
      float p0 = __expf(s0v - m_new), p1 = __expf(s1v - m_new);
      float rs = p0 + p1;
      rs += __shfl_xor(rs, 1, 32);
      rs += __shfl_xor(rs, 2, 32);
      rs += __shfl_xor(rs, 4, 32);
      rs += __shfl_xor(rs, 8, 32);
      l_st[r] = l_st[r]*alpha + rs;
      m_st[r] = m_new;
#pragma unroll
      for (int t = 0; t < 4; ++t) acc[t][r] = acc[t][r] * alpha;
      sP[wave*512 + (r + 8*hi)*32 + ln]      = (_Float16)p0;
      sP[wave*512 + (r + 8*hi)*32 + 16 + ln] = (_Float16)p1;
    }

    // P(16x32) * V(32x64)
    v16h aP = load_frag_a(sP + wave*512, 32);
#pragma unroll
    for (int t = 0; t < 4; ++t) {
      v16h bV = load_frag_b(sVT + t*16*32, 32);
      acc[t] = __builtin_amdgcn_wmma_f32_16x16x32_f16(false, aP, false, bV,
                                                      (short)0, acc[t], false, false);
    }
  }

  // epilogue: ctx[b, s, h*64 + d] = acc / l
#pragma unroll
  for (int t = 0; t < 4; ++t)
#pragma unroll
    for (int r = 0; r < 8; ++r) {
      int qrow = qb + wave*16 + r + 8*hi;
      if (qrow < Ss) {
        float o = acc[t][r] / fmaxf(l_st[r], 1e-20f);
        ctx[((size_t)(b*Ss + qrow))*N_EMBD + h*64 + t*16 + ln] = (_Float16)o;
      }
    }
}

// ---------------- host launcher ----------------
extern "C" void kernel_launch(void* const* d_in, const int* in_sizes, int n_in,
                              void* d_out, int out_size, void* d_ws, size_t ws_size,
                              hipStream_t stream) {
  (void)in_sizes; (void)n_in; (void)out_size; (void)ws_size;
  const float* x   = (const float*)d_in[0];
  const float* qw  = (const float*)d_in[1];
  const float* kw  = (const float*)d_in[2];
  const float* vw  = (const float*)d_in[3];
  const float* qrw = (const float*)d_in[4];
  const float* krw = (const float*)d_in[5];
  const float* ow  = (const float*)d_in[6];
  float* out = (float*)d_out;
  char* ws = (char*)d_ws;

  _Float16* xh   = (_Float16*)(ws + O_XH);
  _Float16* wcat = (_Float16*)(ws + O_WCAT);
  _Float16* owh  = (_Float16*)(ws + O_OWH);
  float*    proj = (float*)   (ws + O_PROJ);
  _Float16* Qb   = (_Float16*)(ws + O_QB);
  _Float16* Kb   = (_Float16*)(ws + O_KB);
  _Float16* VT   = (_Float16*)(ws + O_VT);
  _Float16* ctx  = (_Float16*)(ws + O_CTX);

  pack_f16<<<512, 256, 0, stream>>>(x, qw, kw, vw, qrw, krw, ow, xh, wcat, owh);

  dim3 g1(NPROJ/64, (Mrows + 63)/64);            // 15 x 65
  gemm_wmma_f16<<<g1, 128, 0, stream>>>(xh, wcat, proj, Mrows, NPROJ, N_EMBD);

  finish_q<<<1024, 256, 0, stream>>>(proj, Qb);
  dim3 g2(KQ_LEN/GROUPQ + 1, N_KV, Bb);          // 33 x 3 x 2
  finish_k<<<g2, 64, 0, stream>>>(proj, Kb);
  finish_v<<<128, 256, 0, stream>>>(proj, VT);

  dim3 g3((Ss + 63)/64, Bb*N_HEAD);              // 33 x 18
  attn_wmma<<<g3, 128, 0, stream>>>(Qb, Kb, VT, ctx);

  dim3 g4(N_EMBD/64, (Mrows + 63)/64);           // 9 x 65
  gemm_wmma_f16<<<g4, 128, 0, stream>>>(ctx, owh, out, Mrows, N_EMBD, N_EMBD);
}
